// AdditiveAttention_68092411510948
// MI455X (gfx1250) — matrix operand order repaired and checked by reading, not compiled, not run on hardware
//
#include <hip/hip_runtime.h>
#include <math.h>

#define BQ 4
#define LQ 256
#define LK 512
#define DD 256

typedef __attribute__((ext_vector_type(16))) _Float16 v16h;
typedef __attribute__((ext_vector_type(8)))  float    v8f;

#if __has_builtin(__builtin_amdgcn_tanhf)
__device__ __forceinline__ float fast_tanh(float x) { return __builtin_amdgcn_tanhf(x); }
#else
__device__ __forceinline__ float fast_tanh(float x) { return tanhf(x); }
#endif

__device__ __forceinline__ void wait_asynccnt0() {
#if __has_builtin(__builtin_amdgcn_s_wait_asynccnt)
    __builtin_amdgcn_s_wait_asynccnt(0);
#else
    asm volatile("s_wait_asynccnt 0x0" ::: "memory");
#endif
}

// Async copy of 16 bytes: global -> LDS, tracked by ASYNCcnt (ISA 10.7/15.18).
__device__ __forceinline__ void async_g2l_b128(uint32_t lds_byte_addr, uint64_t gaddr) {
    asm volatile("global_load_async_to_lds_b128 %0, %1, off"
                 :: "v"(lds_byte_addr), "v"(gaddr)
                 : "memory");
}

// A-matrix (16x32) per-lane K mapping, matching the contiguous B loads:
// lanes 0-15 hold row M=lane with K in {0..15}; lanes 16-31 hold the
// same rows with K in {16..31}.
__device__ __forceinline__ int kmapA(int j, int hi) {
    return j + hi * 16;
}

// ---------------------------------------------------------------------------
// Phase 0: f32 -> f16 conversion; also transpose value to [b][d][k] so the
// WMMA B-operand (K-major per lane) loads are contiguous 32B chunks.
// ---------------------------------------------------------------------------
__global__ void convert_kernel(const float* __restrict__ query,
                               const float* __restrict__ key,
                               const float* __restrict__ value,
                               const float* __restrict__ W1,
                               const float* __restrict__ W2,
                               _Float16* __restrict__ keyH,
                               _Float16* __restrict__ queryH,
                               _Float16* __restrict__ W1H,
                               _Float16* __restrict__ W2H,
                               _Float16* __restrict__ valueTH) {
    int i = blockIdx.x * blockDim.x + threadIdx.x;   // grid covers BQ*LK*DD
    if (i < BQ * LK * DD) {
        keyH[i] = (_Float16)key[i];
        int b = i / (LK * DD);
        int r = i % (LK * DD);
        int k = r / DD, d = r % DD;
        valueTH[(size_t)b * DD * LK + (size_t)d * LK + k] = (_Float16)value[i];
    }
    if (i < BQ * LQ * DD) queryH[i] = (_Float16)query[i];
    if (i < DD * DD) {
        W1H[i] = (_Float16)W1[i];
        W2H[i] = (_Float16)W2[i];
    }
}

// ---------------------------------------------------------------------------
// Phase 1: projections via WMMA.
//   W1K[b,k,e] = sum_d key[b,k,d]*W1[e,d]   -> stored transposed: W1Kt[b][e][k]
//   W2Q[b,q,e] = sum_d query[b,q,d]*W2[e,d] -> stored normal:     W2Q[b][q][e]
// One wave per 16x16 output tile; K-loop of 8 chained v_wmma_f32_16x16x32_f16.
// ---------------------------------------------------------------------------
__global__ void proj_kernel(const _Float16* __restrict__ keyH,
                            const _Float16* __restrict__ queryH,
                            const _Float16* __restrict__ W1H,
                            const _Float16* __restrict__ W2H,
                            float* __restrict__ W1Kt,
                            float* __restrict__ W2Q) {
    const int lane = threadIdx.x & 31;
    const int wave = threadIdx.x >> 5;
    const int gid  = blockIdx.x * 8 + wave;          // 3072 tiles total
    const int hi   = lane >> 4;
    const int lo   = lane & 15;

    if (gid < BQ * (LK / 16) * (DD / 16)) {          // 2048 W1K tiles
        const int b  = gid >> 9;
        const int t  = gid & 511;
        const int rt = t >> 4;                       // k row-tile
        const int ct = t & 15;                       // e col-tile
        const _Float16* arow = keyH + ((size_t)b * LK + rt * 16 + lo) * DD;
        const _Float16* brow = W1H + (size_t)(ct * 16 + lo) * DD;   // B col e = W1 row e
        v8f c = {};
        for (int kk = 0; kk < DD / 32; ++kk) {
            const int base = kk * 32;
            v16h a, bv;
#pragma unroll
            for (int j = 0; j < 16; ++j) a[j] = arow[base + kmapA(j, hi)];
            bv = *(const v16h*)(brow + base + hi * 16);
            c = __builtin_amdgcn_wmma_f32_16x16x32_f16(false, a, false, bv,
                                                       (short)0, c, false, false);
        }
        // store transposed [b][e][k]: per lane 8 contiguous k's
        float* out = W1Kt + (size_t)b * DD * LK + (size_t)(ct * 16 + lo) * LK
                   + rt * 16 + hi * 8;
#pragma unroll
        for (int r = 0; r < 8; ++r) out[r] = c[r];
    } else if (gid < 3072) {                          // 1024 W2Q tiles
        const int g  = gid - 2048;
        const int b  = g >> 8;
        const int t  = g & 255;
        const int rt = t >> 4;                       // q row-tile
        const int ct = t & 15;                       // e col-tile
        const _Float16* arow = queryH + ((size_t)b * LQ + rt * 16 + lo) * DD;
        const _Float16* brow = W2H + (size_t)(ct * 16 + lo) * DD;
        v8f c = {};
        for (int kk = 0; kk < DD / 32; ++kk) {
            const int base = kk * 32;
            v16h a, bv;
#pragma unroll
            for (int j = 0; j < 16; ++j) a[j] = arow[base + kmapA(j, hi)];
            bv = *(const v16h*)(brow + base + hi * 16);
            c = __builtin_amdgcn_wmma_f32_16x16x32_f16(false, a, false, bv,
                                                       (short)0, c, false, false);
        }
        float* out = W2Q + (size_t)b * LQ * DD + (size_t)(rt * 16 + hi * 8) * DD
                   + ct * 16 + lo;
#pragma unroll
        for (int r = 0; r < 8; ++r) out[(size_t)r * DD] = c[r];
    }
}

// ---------------------------------------------------------------------------
// Phase 2: fused energy/tanh/score/mask/softmax.
// Block = 256 threads (8 waves) = 4 consecutive q rows x 2 waves (k-halves).
// W1Kt is consumed in 32-row x 512-col f32 chunks (64KB) staged into LDS via
// global_load_async_to_lds_b128 (ASYNCcnt), shared by all 8 waves -> 8x less
// L2 traffic than per-wave streaming. The [B,LQ,LK,D] energy tensor is never
// materialized; each lane keeps 8 k-columns in registers and accumulates the
// v-weighted tanh online (native v_tanh_f32). Cross-wave softmax merge uses
// the exact (mx,sum) rescale identity through 64B of LDS.
// ---------------------------------------------------------------------------
__global__ void score_softmax_kernel(const float* __restrict__ W1Kt,
                                     const float* __restrict__ W2Q,
                                     const float* __restrict__ vvec,
                                     const int* __restrict__ mask,
                                     float* __restrict__ attOut) {
    __shared__ float tile[32 * LK];                  // exactly 64 KB

    const int tid  = threadIdx.x;
    const int lane = tid & 31;
    const int wave = tid >> 5;
    const int b    = blockIdx.x >> 6;                // 64 blocks per batch
    const int q    = ((blockIdx.x & 63) << 2) + (wave >> 1);
    const int half = wave & 1;                       // which 256 k-columns

    float acc[8];
#pragma unroll
    for (int j = 0; j < 8; ++j) acc[j] = 0.0f;

    const float* w2row  = W2Q + ((size_t)b * LQ + q) * DD;
    const float* w1base = W1Kt + (size_t)b * DD * LK;
    const uint32_t lds0 = (uint32_t)(uintptr_t)&tile[0];

    for (int d0 = 0; d0 < DD; d0 += 32) {
        // --- async-stage 32 contiguous rows (64KB) of W1Kt into LDS ---
        const uint64_t g0 = (uint64_t)(uintptr_t)(w1base + (size_t)d0 * LK);
#pragma unroll
        for (int i = 0; i < 16; ++i) {
            const uint32_t off = (uint32_t)(tid + 256 * i) * 16u;  // float4 units
            async_g2l_b128(lds0 + off, g0 + off);
        }
        // L2 prefetch of the next chunk while this one lands
        if (d0 + 32 < DD)
            __builtin_prefetch(w1base + (size_t)(d0 + 32) * LK + tid * 64, 0, 0);
        wait_asynccnt0();
        __syncthreads();

        // --- consume: 32 d-rows x 8 k-regs per lane ---
        const float w2v = w2row[d0 + lane];
        const float vv  = vvec[d0 + lane];
        for (int t = 0; t < 32; ++t) {
            const float w2t = __shfl(w2v, t, 32);
            const float vt  = __shfl(vv, t, 32);
            const float* row = tile + t * LK + half * 256;
#pragma unroll
            for (int j = 0; j < 8; ++j)
                acc[j] = fmaf(vt, fast_tanh(row[j * 32 + lane] + w2t), acc[j]);
        }
        __syncthreads();   // before next chunk overwrites the tile
    }

    // scale (1/sqrt(256)), mask, per-wave softmax stats over this half (256 vals)
    const int* mrow = mask + ((size_t)b * LQ + q) * LK;
    float s[8];
    float mx = -3.0e38f;
#pragma unroll
    for (int j = 0; j < 8; ++j) {
        const int k = half * 256 + j * 32 + lane;
        float sc = acc[j] * 0.0625f;
        if (mrow[k] == 0) sc = -1.0e10f;
        s[j] = sc;
        mx = fmaxf(mx, sc);
    }
#pragma unroll
    for (int off = 16; off > 0; off >>= 1) mx = fmaxf(mx, __shfl_xor(mx, off, 32));
    float sum = 0.0f;
#pragma unroll
    for (int j = 0; j < 8; ++j) sum += __expf(s[j] - mx);
#pragma unroll
    for (int off = 16; off > 0; off >>= 1) sum += __shfl_xor(sum, off, 32);

    // merge the two k-halves of each q (exact rescale), via 64B of LDS scratch
    if (lane == 0) { tile[wave] = mx; tile[8 + wave] = sum; }
    __syncthreads();
    const int   partner = wave ^ 1;
    const float pmx = tile[partner], psum = tile[8 + partner];
    const float M = fmaxf(mx, pmx);
    const float S = sum * __expf(mx - M) + psum * __expf(pmx - M);
    const float inv = 1.0f / S;

    float* arow = attOut + ((size_t)b * LQ + q) * LK;
#pragma unroll
    for (int j = 0; j < 8; ++j) {
        const int k = half * 256 + j * 32 + lane;
        arow[k] = __expf(s[j] - M) * inv;
    }
}

// ---------------------------------------------------------------------------
// Phase 3: context = attention x value via WMMA, one wave per 16x16 tile,
// K-loop of 16 chained v_wmma_f32_16x16x32_f16 over LK=512.
// B-operand reads the pre-transposed f16 value (contiguous per lane).
// ---------------------------------------------------------------------------
__global__ void context_kernel(const float* __restrict__ att,
                               const _Float16* __restrict__ valueTH,
                               float* __restrict__ ctx) {
    const int lane = threadIdx.x & 31;
    const int wave = threadIdx.x >> 5;
    const int gid  = blockIdx.x * 8 + wave;          // 1024 tiles
    const int hi   = lane >> 4;
    const int lo   = lane & 15;
    const int b  = gid >> 8;
    const int t  = gid & 255;
    const int qt = t >> 4;
    const int dt = t & 15;

    const float*    arow = att + ((size_t)b * LQ + qt * 16 + lo) * LK;
    const _Float16* brow = valueTH + (size_t)b * DD * LK + (size_t)(dt * 16 + lo) * LK;

    v8f c = {};
    for (int kk = 0; kk < LK / 32; ++kk) {
        const int base = kk * 32;
        v16h a, bv;
#pragma unroll
        for (int j = 0; j < 16; ++j) a[j] = (_Float16)arow[base + kmapA(j, hi)];
        bv = *(const v16h*)(brow + base + hi * 16);
        c = __builtin_amdgcn_wmma_f32_16x16x32_f16(false, a, false, bv,
                                                   (short)0, c, false, false);
    }
    float* orow = ctx + (size_t)b * LQ * DD + (size_t)(qt * 16 + hi * 8) * DD
                + dt * 16 + lo;
#pragma unroll
    for (int r = 0; r < 8; ++r) orow[(size_t)r * DD] = c[r];
}

// ---------------------------------------------------------------------------
extern "C" void kernel_launch(void* const* d_in, const int* in_sizes, int n_in,
                              void* d_out, int out_size, void* d_ws, size_t ws_size,
                              hipStream_t stream) {
    (void)in_sizes; (void)n_in; (void)out_size; (void)ws_size;
    const float* query = (const float*)d_in[0];
    const float* key   = (const float*)d_in[1];
    const float* value = (const float*)d_in[2];
    const float* W1    = (const float*)d_in[3];
    const float* W2    = (const float*)d_in[4];
    const float* v     = (const float*)d_in[5];
    const int*   mask  = (const int*)d_in[6];

    char* ws = (char*)d_ws;
    _Float16* keyH    = (_Float16*)(ws);              // 1,048,576 B
    _Float16* queryH  = (_Float16*)(ws + 1048576);    //   524,288 B
    _Float16* W1H     = (_Float16*)(ws + 1572864);    //   131,072 B
    _Float16* W2H     = (_Float16*)(ws + 1703936);    //   131,072 B
    _Float16* valueTH = (_Float16*)(ws + 1835008);    // 1,048,576 B  [b][d][k]
    float*    W1Kt    = (float*)   (ws + 2883584);    // 2,097,152 B  [b][d][k]
    float*    W2Q     = (float*)   (ws + 4980736);    // 1,048,576 B  [b][q][d]

    float* ctx = (float*)d_out;                        // [B,LQ,D]
    float* att = (float*)d_out + (size_t)BQ * LQ * DD; // [B,LQ,LK]

    convert_kernel<<<2048, 256, 0, stream>>>(query, key, value, W1, W2,
                                             keyH, queryH, W1H, W2H, valueTH);
    proj_kernel<<<384, 256, 0, stream>>>(keyH, queryH, W1H, W2H, W1Kt, W2Q);
    score_softmax_kernel<<<256, 256, 0, stream>>>(W1Kt, W2Q, v, mask, att);
    context_kernel<<<128, 256, 0, stream>>>(att, valueTH, ctx);
}
